// FFFN_47296179864206
// MI455X (gfx1250) — matrix operand order repaired and checked by reading, not compiled, not run on hardware
//
#include <hip/hip_runtime.h>
#include <hip/hip_bf16.h>
#include <math.h>
#include <stdint.h>

// ---------------- problem constants ----------------
#define NTOK   2048      // B*S
#define NEXP   4096      // NTOK * K
#define DMODEL 1024
#define DHID   4096
#define NE     8
#define LN_EPS 1e-5f

// block tiling
#define BM 128
#define BN 128
#define BK 32
#define LDS_STRIDE 40    // 32 + 8 pad (bf16 units): 80B row stride, conflict-free

typedef __attribute__((ext_vector_type(16))) __bf16 v16bf;
typedef __attribute__((ext_vector_type(8)))  __bf16 v8bf;
typedef __attribute__((ext_vector_type(8)))  float  v8f;
typedef __attribute__((ext_vector_type(4)))  float  v4f;

// ---------------- workspace layout (bytes) ----------------
#define WS_COUNTS 0
#define WS_CTR    64
#define WS_OFFS   128
#define WS_EID    256
#define WS_BTOK   (WS_EID  + 4096*4)
#define WS_BDST   (WS_BTOK + 4096*4)
#define WS_SCORE  (WS_BDST + 4096*4)
#define WS_HBUF   (1u<<20)
#define WS_YEXP   (WS_HBUF + (size_t)NEXP*DHID*2)

__device__ __forceinline__ float gelu_exact(float x) {
    return 0.5f * x * (1.0f + erff(x * 0.70710678118654752f));
}

// pack 8 fp32 (two v4f) into v8bf
__device__ __forceinline__ v8bf pack_bf8(v4f a, v4f b) {
    v8bf r;
#pragma unroll
    for (int i = 0; i < 4; ++i) { r[i] = (__bf16)a[i]; r[4 + i] = (__bf16)b[i]; }
    return r;
}

// ---- CDNA5 async global->LDS copy (ASYNCcnt path), inline asm ----
// LDS destination = low 32 bits of the generic shared-memory address
// (ISA: LDS aperture addresses truncate to the LDS byte offset).
__device__ __forceinline__ void async_copy_b128(void* lds_dst, const void* gsrc) {
    uint32_t ldsoff = (uint32_t)(uintptr_t)lds_dst;
    asm volatile("global_load_async_to_lds_b128 %0, %1, off"
                 :: "v"(ldsoff), "v"(gsrc) : "memory");
}
__device__ __forceinline__ void wait_async0() {
    asm volatile("s_wait_asynccnt 0x0" ::: "memory");
}

// ================= kernel 0: zero counters =================
__global__ void k_zero(int* counts, int* ctr) {
    int i = threadIdx.x;
    if (i < NE) { counts[i] = 0; ctr[i] = 0; }
}

// ================= kernel 1: router (one wave per token) =================
__global__ __launch_bounds__(256)
void k_router(const float* __restrict__ x, const float* __restrict__ gw,
              const float* __restrict__ gb,
              int* __restrict__ eid, float* __restrict__ score,
              int* __restrict__ counts) {
    const int wave = threadIdx.x >> 5;
    const int lane = threadIdx.x & 31;
    const int t = blockIdx.x * 8 + wave;
    if (t >= NTOK) return;

    const float* xr = x + (size_t)t * DMODEL;
    float acc[NE];
#pragma unroll
    for (int e = 0; e < NE; ++e) acc[e] = 0.0f;

    for (int d = lane; d < DMODEL; d += 32) {
        float xv = xr[d];
#pragma unroll
        for (int e = 0; e < NE; ++e) acc[e] += xv * gw[e * DMODEL + d];
    }
#pragma unroll
    for (int e = 0; e < NE; ++e) {
#pragma unroll
        for (int off = 16; off > 0; off >>= 1)
            acc[e] += __shfl_xor(acc[e], off, 32);
    }
    if (lane == 0) {
#pragma unroll
        for (int e = 0; e < NE; ++e) acc[e] += gb[e];
        float v0 = -INFINITY; int e0 = 0;
#pragma unroll
        for (int e = 0; e < NE; ++e) if (acc[e] > v0) { v0 = acc[e]; e0 = e; }
        float v1 = -INFINITY; int e1 = 0;
#pragma unroll
        for (int e = 0; e < NE; ++e) if (e != e0 && acc[e] > v1) { v1 = acc[e]; e1 = e; }
        float p0 = 1.0f / (1.0f + expf(v1 - v0));   // softmax over top-2
        float p1 = 1.0f - p0;
        eid[2 * t]     = e0;  score[2 * t]     = p0;
        eid[2 * t + 1] = e1;  score[2 * t + 1] = p1;
        atomicAdd(&counts[e0], 1);
        atomicAdd(&counts[e1], 1);
    }
}

// ================= kernel 2: prefix sum over 8 experts =================
__global__ void k_prefix(const int* __restrict__ counts, int* __restrict__ offs) {
    if (threadIdx.x == 0) {
        int s = 0;
        for (int e = 0; e < NE; ++e) { offs[e] = s; s += counts[e]; }
    }
}

// ================= kernel 3: scatter expanded tokens into buckets =================
__global__ void k_scatter(const int* __restrict__ eid, const int* __restrict__ offs,
                          int* __restrict__ ctr,
                          int* __restrict__ btok, int* __restrict__ bdst) {
    int j = blockIdx.x * blockDim.x + threadIdx.x;
    if (j >= NEXP) return;
    int e = eid[j];
    int pos = offs[e] + atomicAdd(&ctr[e], 1);
    btok[pos] = j >> 1;   // source token
    bdst[pos] = j;        // expanded destination row
}

// ================= GEMM kernels (LDS-staged, double buffered) =================
// Block: 256 threads = 8 wave32; tile 128x128, K step 32 (bf16 WMMA).
// Fragment reads from LDS (row stride LDS_STRIDE bf16):
//  A (16x32, lane L): row = L%16; chunks at k = (L/16)*8 and 16+(L/16)*8 (8 bf16 each)
//  B (32x16, lane L): col-row = L%16; chunks at k = (L/16)*8 and 16+(L/16)*8 (8 bf16 each)
//  C frag: VGPR v, lane L: M = (L/16)*8 + v, N = L%16.

__global__ __launch_bounds__(256)
void k_gemm1(const float* __restrict__ x, const float* __restrict__ w1,
             const int* __restrict__ btok, const int* __restrict__ counts,
             const int* __restrict__ offs, __bf16* __restrict__ Hbuf) {
    __shared__ __bf16 As[2][BM * LDS_STRIDE];
    __shared__ __bf16 Bs[2][BN * LDS_STRIDE];

    const int e     = blockIdx.z;
    const int mtile = blockIdx.y;
    const int ntile = blockIdx.x;
    const int cnt   = counts[e];
    if (mtile * BM >= cnt) return;
    const int base  = offs[e];

    const int tid  = threadIdx.x;
    const int lane = tid & 31;
    const int wave = tid >> 5;
    const int wm   = wave & 1;        // 0..1  (M)
    const int wn   = wave >> 1;       // 0..3  (N)
    const int half = lane >> 4;       // 0..1
    const int r    = lane & 15;

    // staging coordinates
    const int srow = tid >> 1;        // 0..127
    const int sseg = tid & 1;         // 0..1
    const int soff = srow * LDS_STRIDE + sseg * 16;

    int mrow0 = mtile * BM + srow;
    int mcl   = (mrow0 < cnt) ? mrow0 : (cnt - 1);
    const float* agp = x + (size_t)btok[base + mcl] * DMODEL + sseg * 16;
    const float* bgp = w1 + (size_t)e * DHID * DMODEL
                          + (size_t)(ntile * BN + srow) * DMODEL + sseg * 16;

    v4f ra0 = *(const v4f*)(agp + 0), ra1 = *(const v4f*)(agp + 4);
    v4f ra2 = *(const v4f*)(agp + 8), ra3 = *(const v4f*)(agp + 12);
    v4f rb0 = *(const v4f*)(bgp + 0), rb1 = *(const v4f*)(bgp + 4);
    v4f rb2 = *(const v4f*)(bgp + 8), rb3 = *(const v4f*)(bgp + 12);

    v8f acc[4][2];
    v8f zf = {0.f,0.f,0.f,0.f,0.f,0.f,0.f,0.f};
#pragma unroll
    for (int ti = 0; ti < 4; ++ti)
#pragma unroll
        for (int tj = 0; tj < 2; ++tj) acc[ti][tj] = zf;

    int buf = 0;
    for (int k0 = 0; k0 < DMODEL; k0 += BK, buf ^= 1) {
        // convert + stage current regs into LDS[buf]
        *(v8bf*)&As[buf][soff]     = pack_bf8(ra0, ra1);
        *(v8bf*)&As[buf][soff + 8] = pack_bf8(ra2, ra3);
        *(v8bf*)&Bs[buf][soff]     = pack_bf8(rb0, rb1);
        *(v8bf*)&Bs[buf][soff + 8] = pack_bf8(rb2, rb3);
        __syncthreads();

        // prefetch next k-step while WMMAs run
        if (k0 + BK < DMODEL) {
            int kn = k0 + BK;
            ra0 = *(const v4f*)(agp + kn);     ra1 = *(const v4f*)(agp + kn + 4);
            ra2 = *(const v4f*)(agp + kn + 8); ra3 = *(const v4f*)(agp + kn + 12);
            rb0 = *(const v4f*)(bgp + kn);     rb1 = *(const v4f*)(bgp + kn + 4);
            rb2 = *(const v4f*)(bgp + kn + 8); rb3 = *(const v4f*)(bgp + kn + 12);
        }

        v16bf a[4];
#pragma unroll
        for (int ti = 0; ti < 4; ++ti) {
            const __bf16* p = &As[buf][(wm * 64 + ti * 16 + r) * LDS_STRIDE + half * 8];
            v8bf c0 = *(const v8bf*)(p);
            v8bf c1 = *(const v8bf*)(p + 16);
#pragma unroll
            for (int i = 0; i < 8; ++i) { a[ti][i] = c0[i]; a[ti][8 + i] = c1[i]; }
        }
        v16bf b[2];
#pragma unroll
        for (int tj = 0; tj < 2; ++tj) {
            const __bf16* p = &Bs[buf][(wn * 32 + tj * 16 + r) * LDS_STRIDE + half * 8];
            v8bf c0 = *(const v8bf*)(p);
            v8bf c1 = *(const v8bf*)(p + 16);
#pragma unroll
            for (int i = 0; i < 8; ++i) { b[tj][i] = c0[i]; b[tj][8 + i] = c1[i]; }
        }
#pragma unroll
        for (int ti = 0; ti < 4; ++ti)
#pragma unroll
            for (int tj = 0; tj < 2; ++tj)
                acc[ti][tj] = __builtin_amdgcn_wmma_f32_16x16x32_bf16(
                    false, a[ti], false, b[tj], (short)0, acc[ti][tj], false, false);
    }

    // gelu + store bf16 hidden activations (half-wave writes 32 contiguous bytes)
#pragma unroll
    for (int ti = 0; ti < 4; ++ti) {
#pragma unroll
        for (int tj = 0; tj < 2; ++tj) {
            int col = ntile * BN + wn * 32 + tj * 16 + r;
#pragma unroll
            for (int v = 0; v < 8; ++v) {
                int mrow = mtile * BM + wm * 64 + ti * 16 + half * 8 + v;
                if (mrow < cnt) {
                    float g = gelu_exact(acc[ti][tj][v]);
                    Hbuf[(size_t)(base + mrow) * DHID + col] = (__bf16)g;
                }
            }
        }
    }
}

__global__ __launch_bounds__(256)
void k_gemm2(const __bf16* __restrict__ Hbuf, const float* __restrict__ w2,
             const int* __restrict__ bdst, const int* __restrict__ counts,
             const int* __restrict__ offs, float* __restrict__ Yexp) {
    __shared__ __bf16 As[2][BM * LDS_STRIDE];
    __shared__ __bf16 Bs[2][BN * LDS_STRIDE];

    const int e     = blockIdx.z;
    const int mtile = blockIdx.y;
    const int ntile = blockIdx.x;
    const int cnt   = counts[e];
    if (mtile * BM >= cnt) return;
    const int base  = offs[e];

    const int tid  = threadIdx.x;
    const int lane = tid & 31;
    const int wave = tid >> 5;
    const int wm   = wave & 1;
    const int wn   = wave >> 1;
    const int half = lane >> 4;
    const int r    = lane & 15;

    const int srow = tid >> 1;
    const int sseg = tid & 1;
    const int soff = srow * LDS_STRIDE + sseg * 16;

    int mrow0 = mtile * BM + srow;
    int mcl   = (mrow0 < cnt) ? mrow0 : (cnt - 1);
    const __bf16* agp = Hbuf + (size_t)(base + mcl) * DHID + sseg * 16;
    const float*  bgp = w2 + (size_t)e * DMODEL * DHID
                           + (size_t)(ntile * BN + srow) * DHID + sseg * 16;

    // A tile (bf16 -> bf16, pure copy): stream via async global->LDS (ASYNCcnt).
    // Prologue: issue tile 0 into LDS buffer 0.
    async_copy_b128(&As[0][soff],     agp);
    async_copy_b128(&As[0][soff + 8], agp + 8);

    // B tile (fp32 -> bf16): register-staged + converted.
    v4f  rb0 = *(const v4f*)(bgp + 0),  rb1 = *(const v4f*)(bgp + 4);
    v4f  rb2 = *(const v4f*)(bgp + 8),  rb3 = *(const v4f*)(bgp + 12);

    v8f acc[4][2];
    v8f zf = {0.f,0.f,0.f,0.f,0.f,0.f,0.f,0.f};
#pragma unroll
    for (int ti = 0; ti < 4; ++ti)
#pragma unroll
        for (int tj = 0; tj < 2; ++tj) acc[ti][tj] = zf;

    int buf = 0;
    for (int k0 = 0; k0 < DHID; k0 += BK, buf ^= 1) {
        *(v8bf*)&Bs[buf][soff]     = pack_bf8(rb0, rb1);
        *(v8bf*)&Bs[buf][soff + 8] = pack_bf8(rb2, rb3);
        wait_async0();            // A tile for this buffer has landed in LDS
        __syncthreads();

        if (k0 + BK < DHID) {
            int kn = k0 + BK;
            // next A tile -> other LDS buffer, fully asynchronous
            async_copy_b128(&As[buf ^ 1][soff],     agp + kn);
            async_copy_b128(&As[buf ^ 1][soff + 8], agp + kn + 8);
            rb0 = *(const v4f*)(bgp + kn);     rb1 = *(const v4f*)(bgp + kn + 4);
            rb2 = *(const v4f*)(bgp + kn + 8); rb3 = *(const v4f*)(bgp + kn + 12);
        }

        v16bf a[4];
#pragma unroll
        for (int ti = 0; ti < 4; ++ti) {
            const __bf16* p = &As[buf][(wm * 64 + ti * 16 + r) * LDS_STRIDE + half * 8];
            v8bf c0 = *(const v8bf*)(p);
            v8bf c1 = *(const v8bf*)(p + 16);
#pragma unroll
            for (int i = 0; i < 8; ++i) { a[ti][i] = c0[i]; a[ti][8 + i] = c1[i]; }
        }
        v16bf b[2];
#pragma unroll
        for (int tj = 0; tj < 2; ++tj) {
            const __bf16* p = &Bs[buf][(wn * 32 + tj * 16 + r) * LDS_STRIDE + half * 8];
            v8bf c0 = *(const v8bf*)(p);
            v8bf c1 = *(const v8bf*)(p + 16);
#pragma unroll
            for (int i = 0; i < 8; ++i) { b[tj][i] = c0[i]; b[tj][8 + i] = c1[i]; }
        }
#pragma unroll
        for (int ti = 0; ti < 4; ++ti)
#pragma unroll
            for (int tj = 0; tj < 2; ++tj)
                acc[ti][tj] = __builtin_amdgcn_wmma_f32_16x16x32_bf16(
                    false, a[ti], false, b[tj], (short)0, acc[ti][tj], false, false);
    }

    // scatter rows to expanded-token destinations (half-wave writes 64B runs)
#pragma unroll
    for (int ti = 0; ti < 4; ++ti) {
#pragma unroll
        for (int tj = 0; tj < 2; ++tj) {
            int col = ntile * BN + wn * 32 + tj * 16 + r;
#pragma unroll
            for (int v = 0; v < 8; ++v) {
                int mrow = mtile * BM + wm * 64 + ti * 16 + half * 8 + v;
                if (mrow < cnt) {
                    int j = bdst[base + mrow];
                    Yexp[(size_t)j * DMODEL + col] = acc[ti][tj][v];
                }
            }
        }
    }
}

// ================= kernel 6: combine + residual + LayerNorm =================
__global__ __launch_bounds__(256)
void k_combine_ln(const float* __restrict__ x, const float* __restrict__ Yexp,
                  const float* __restrict__ score,
                  const float* __restrict__ gamma, const float* __restrict__ beta,
                  float* __restrict__ out) {
    const int t   = blockIdx.x;
    const int tid = threadIdx.x;
    __shared__ float ssum[8], ssq[8];

    float s0 = score[2 * t], s1 = score[2 * t + 1];
    const float* xr = x + (size_t)t * DMODEL;
    const float* y0 = Yexp + (size_t)(2 * t) * DMODEL;
    const float* y1 = y0 + DMODEL;

    float v[4], sum = 0.f, sq = 0.f;
#pragma unroll
    for (int i = 0; i < 4; ++i) {
        int d = tid + i * 256;
        float o = xr[d] + s0 * y0[d] + s1 * y1[d];
        v[i] = o; sum += o; sq += o * o;
    }
#pragma unroll
    for (int off = 16; off > 0; off >>= 1) {
        sum += __shfl_xor(sum, off, 32);
        sq  += __shfl_xor(sq,  off, 32);
    }
    int w = tid >> 5, ln = tid & 31;
    if (ln == 0) { ssum[w] = sum; ssq[w] = sq; }
    __syncthreads();
    if (w == 0) {
        float a = (ln < 8) ? ssum[ln] : 0.f;
        float b = (ln < 8) ? ssq[ln]  : 0.f;
#pragma unroll
        for (int off = 4; off > 0; off >>= 1) {
            a += __shfl_xor(a, off, 32);
            b += __shfl_xor(b, off, 32);
        }
        if (ln == 0) { ssum[0] = a; ssq[0] = b; }
    }
    __syncthreads();
    float mu  = ssum[0] * (1.0f / DMODEL);
    float var = ssq[0] * (1.0f / DMODEL) - mu * mu;
    float rs  = rsqrtf(var + LN_EPS);
#pragma unroll
    for (int i = 0; i < 4; ++i) {
        int d = tid + i * 256;
        out[(size_t)t * DMODEL + d] = (v[i] - mu) * rs * gamma[d] + beta[d];
    }
}

// ================= kernel 7: bias passthrough (2nd tuple element) =================
__global__ void k_bias(const float* __restrict__ bias, float* __restrict__ out) {
    int i = blockIdx.x * blockDim.x + threadIdx.x;
    if (i < DMODEL) out[(size_t)NTOK * DMODEL + i] = bias[i];
}

// ================= launcher =================
extern "C" void kernel_launch(void* const* d_in, const int* in_sizes, int n_in,
                              void* d_out, int out_size, void* d_ws, size_t ws_size,
                              hipStream_t stream) {
    const float* x      = (const float*)d_in[0];
    const float* gate_w = (const float*)d_in[1];
    const float* gate_b = (const float*)d_in[2];
    const float* w1     = (const float*)d_in[3];
    const float* w2     = (const float*)d_in[4];
    const float* ln_g   = (const float*)d_in[5];
    const float* ln_b   = (const float*)d_in[6];
    const float* bias   = (const float*)d_in[7];
    float* out = (float*)d_out;

    char* ws = (char*)d_ws;
    int*    counts = (int*)(ws + WS_COUNTS);
    int*    ctr    = (int*)(ws + WS_CTR);
    int*    offs   = (int*)(ws + WS_OFFS);
    int*    eid    = (int*)(ws + WS_EID);
    int*    btok   = (int*)(ws + WS_BTOK);
    int*    bdst   = (int*)(ws + WS_BDST);
    float*  score  = (float*)(ws + WS_SCORE);
    __bf16* Hbuf   = (__bf16*)(ws + WS_HBUF);
    float*  Yexp   = (float*)(ws + WS_YEXP);

    k_zero<<<1, 64, 0, stream>>>(counts, ctr);
    k_router<<<NTOK / 8, 256, 0, stream>>>(x, gate_w, gate_b, eid, score, counts);
    k_prefix<<<1, 32, 0, stream>>>(counts, offs);
    k_scatter<<<NEXP / 256, 256, 0, stream>>>(eid, offs, ctr, btok, bdst);

    k_gemm1<<<dim3(DHID / BN, NEXP / BM, NE), 256, 0, stream>>>(
        x, w1, btok, counts, offs, Hbuf);
    k_gemm2<<<dim3(DMODEL / BN, NEXP / BM, NE), 256, 0, stream>>>(
        Hbuf, w2, bdst, counts, offs, Yexp);

    k_combine_ln<<<NTOK, 256, 0, stream>>>(x, Yexp, score, ln_g, ln_b, out);
    k_bias<<<(DMODEL + 255) / 256, 256, 0, stream>>>(bias, out);
}
